// SS2D_64183991272092
// MI455X (gfx1250) — compile-verified, matrix operand order checked by reading, not run on hardware
//
#include <hip/hip_runtime.h>

// ---------------------------------------------------------------------------
// SS2D (VMamba 4-direction selective scan) for gfx1250 (MI455X).
// GEMMs: v_wmma_f32_16x16x32_bf16, macro-tile 128x64, wave tile 32x32,
// K-contiguous LDS layout for both operands so each fragment is two
// ds_load_b128. Fast-path staging uses f32x4 global loads + packed bf16x8
// ds_store_b128 + next-tile global_prefetch. Scan: 2-pass chunked algorithm
// exploiting dA = exp(dt*A) => chunk carry factor exp(A * sum(dt)).
// ---------------------------------------------------------------------------

typedef __bf16 bf16;
typedef __attribute__((ext_vector_type(16))) __bf16 v16bf;
typedef __attribute__((ext_vector_type(8)))  __bf16 bf16x8;
typedef __attribute__((ext_vector_type(8)))  float  v8f;
typedef __attribute__((ext_vector_type(4)))  float  f32x4;

#define BATCH   2
#define LSEQ    4096          // H*W (64x64)
#define DMODEL  192
#define DINNER  384
#define DSTATE  16
#define DTRANK  24
#define NSEQ    8             // 4 directions * 2 batches
#define NCHUNK  8
#define TCHUNK  (LSEQ / NCHUNK)   // 512

#define TILE_M 128
#define TILE_N 64
#define TILE_K 32
#define KPAD   40             // LDS row pitch in bf16 (80B, keeps 16B alignment)

__device__ __forceinline__ float sigmoidf_(float v) {
    return 1.0f / (1.0f + __expf(-v));
}
__device__ __forceinline__ float softplusf_(float v) {
    return fmaxf(v, 0.0f) + log1pf(__expf(-fabsf(v)));
}

// Build one 16-bit WMMA fragment from a K-contiguous LDS row.
// CDNA5 layout: element i -> k = (i&7) + (i>=8 ? 16 : 0) + kh*8.
__device__ __forceinline__ v16bf frag_k32(const bf16* rowp, int kh) {
    bf16x8 lo = *(const bf16x8*)(rowp + kh * 8);
    bf16x8 hi = *(const bf16x8*)(rowp + 16 + kh * 8);
    return __builtin_shufflevector(lo, hi, 0, 1, 2, 3, 4, 5, 6, 7,
                                           8, 9, 10, 11, 12, 13, 14, 15);
}

__device__ __forceinline__ bf16x8 pack8(f32x4 a, f32x4 b) {
    bf16x8 r;
    #pragma unroll
    for (int i = 0; i < 4; ++i) { r[i] = (bf16)a[i]; r[4 + i] = (bf16)b[i]; }
    return r;
}

// ---------------------------------------------------------------------------
// C[M,N] = A[M,K] * B[K,N] (+bias) (+residual), f32 in/out, bf16 WMMA inside.
// Block: 256 threads (8 waves). Waves: 4 (M) x 2 (N), each owns 32x32.
// ---------------------------------------------------------------------------
__global__ __launch_bounds__(256)
void gemm_wmma_bf16(const float* __restrict__ A, const float* __restrict__ B,
                    float* __restrict__ C,
                    const float* __restrict__ bias,       // nullable, len N
                    const float* __restrict__ residual,   // nullable, ld == ldc
                    int M, int N, int K, int lda, int ldb, int ldc)
{
    __shared__ alignas(16) bf16 As[TILE_M][KPAD];   // [row][k]
    __shared__ alignas(16) bf16 Bs[TILE_N][KPAD];   // [col][k]  (transposed)

    const int tid  = threadIdx.x;
    const int wave = tid >> 5;
    const int lane = tid & 31;
    const int wm   = wave >> 1;        // 0..3 : 32-row band
    const int wn   = wave & 1;         // 0..1 : 32-col band
    const int row0 = blockIdx.y * TILE_M;
    const int col0 = blockIdx.x * TILE_N;

    const int m16 = lane & 15;
    const int kh  = lane >> 4;

    // staging coordinates
    const int ar = tid >> 1;                 // 0..127 (A row)
    const int ak = (tid & 1) * 16;           // 0 / 16 (A k-offset)
    const int bn = tid & 63;                 // 0..63  (B col)
    const int bk = (tid >> 6) * 8;           // 0,8,16,24 (B k-offset)

    const bool interior = (row0 + TILE_M <= M) && (col0 + TILE_N <= N);

    v8f c00 = {}, c01 = {}, c10 = {}, c11 = {};

    for (int k0 = 0; k0 < K; k0 += TILE_K) {
        const bool fullK = (k0 + TILE_K <= K);
        if (interior && fullK) {
            // ---- fast path: vectorized stage, f32 -> bf16 pack ----
            {
                const f32x4* ap =
                    (const f32x4*)(A + (size_t)(row0 + ar) * lda + k0 + ak);
                f32x4 v0 = ap[0], v1 = ap[1], v2 = ap[2], v3 = ap[3];
                *(bf16x8*)&As[ar][ak]     = pack8(v0, v1);
                *(bf16x8*)&As[ar][ak + 8] = pack8(v2, v3);
                if (k0 + TILE_K < K)
                    __builtin_prefetch(A + (size_t)(row0 + ar) * lda +
                                       k0 + TILE_K + ak, 0, 1);
            }
            {
                const float* bp = B + (size_t)(k0 + bk) * ldb + col0 + bn;
                bf16x8 p;
                #pragma unroll
                for (int j = 0; j < 8; ++j) p[j] = (bf16)bp[(size_t)j * ldb];
                *(bf16x8*)&Bs[bn][bk] = p;
                if (k0 + TILE_K < K)
                    __builtin_prefetch(B + (size_t)(k0 + TILE_K + bk) * ldb +
                                       col0 + bn, 0, 1);
            }
        } else {
            // ---- slow path: fully guarded scalar stage (edge tiles only) ----
            {
                const int grow = row0 + ar;
                #pragma unroll
                for (int j = 0; j < 16; ++j) {
                    const int gk = k0 + ak + j;
                    float v = 0.0f;
                    if (grow < M && gk < K) v = A[(size_t)grow * lda + gk];
                    As[ar][ak + j] = (bf16)v;
                }
            }
            {
                const int gc = col0 + bn;
                #pragma unroll
                for (int j = 0; j < 8; ++j) {
                    const int gk = k0 + bk + j;
                    float v = 0.0f;
                    if (gc < N && gk < K) v = B[(size_t)gk * ldb + gc];
                    Bs[bn][bk + j] = (bf16)v;
                }
            }
        }
        __syncthreads();

        const v16bf a0 = frag_k32(&As[wm * 32 + m16][0], kh);
        const v16bf a1 = frag_k32(&As[wm * 32 + 16 + m16][0], kh);
        const v16bf b0 = frag_k32(&Bs[wn * 32 + m16][0], kh);
        const v16bf b1 = frag_k32(&Bs[wn * 32 + 16 + m16][0], kh);

        c00 = __builtin_amdgcn_wmma_f32_16x16x32_bf16(false, a0, false, b0,
                                                      (short)0, c00, false, false);
        c01 = __builtin_amdgcn_wmma_f32_16x16x32_bf16(false, a0, false, b1,
                                                      (short)0, c01, false, false);
        c10 = __builtin_amdgcn_wmma_f32_16x16x32_bf16(false, a1, false, b0,
                                                      (short)0, c10, false, false);
        c11 = __builtin_amdgcn_wmma_f32_16x16x32_bf16(false, a1, false, b1,
                                                      (short)0, c11, false, false);
        __syncthreads();
    }

    // ---- epilogue: VGPR r of tile (ti,tj) -> row wm*32+ti*16+kh*8+r,
    //      col col0 + wn*32 + tj*16 + m16
    auto store_tile = [&](const v8f& c, int trow, int tcol) {
        const int gc = col0 + wn * 32 + tcol + m16;
        #pragma unroll
        for (int r = 0; r < 8; ++r) {
            const int grow = row0 + wm * 32 + trow + kh * 8 + r;
            if (grow < M && gc < N) {
                float v = c[r];
                if (bias)     v += bias[gc];
                if (residual) v += residual[(size_t)grow * ldc + gc];
                C[(size_t)grow * ldc + gc] = v;
            }
        }
    };
    store_tile(c00, 0, 0);
    store_tile(c01, 0, 16);
    store_tile(c10, 16, 0);
    store_tile(c11, 16, 16);
}

// ---------------------------------------------------------------------------
// Build the 4 directional sequences from xs (= cols [0,384) of xz).
// ---------------------------------------------------------------------------
__global__ __launch_bounds__(256)
void build_seqs(const float* __restrict__ xz, float* __restrict__ seqs)
{
    const size_t total = (size_t)NSEQ * LSEQ * DINNER;
    size_t idx = (size_t)blockIdx.x * blockDim.x + threadIdx.x;
    if (idx >= total) return;
    const int d = (int)(idx % DINNER);
    const int t = (int)((idx / DINNER) % LSEQ);
    const int s = (int)(idx / ((size_t)DINNER * LSEQ));
    const int b = s & 1;
    const int dir = s >> 1;
    int l;
    if (dir == 0)      l = t;
    else if (dir == 1) l = LSEQ - 1 - t;
    else if (dir == 2) l = (t & 63) * 64 + (t >> 6);
    else { const int tt = LSEQ - 1 - t; l = (tt & 63) * 64 + (tt >> 6); }
    seqs[idx] = xz[((size_t)b * LSEQ + l) * (2 * DINNER) + d];
}

// ---------------------------------------------------------------------------
// Depthwise causal conv (k=4) + bias + SiLU over xi = cols [0,384) of xz2.
// ---------------------------------------------------------------------------
__global__ __launch_bounds__(256)
void conv_silu(const float* __restrict__ xz2, const float* __restrict__ conv_w,
               const float* __restrict__ conv_b, float* __restrict__ xc)
{
    const size_t total = (size_t)NSEQ * LSEQ * DINNER;
    size_t idx = (size_t)blockIdx.x * blockDim.x + threadIdx.x;
    if (idx >= total) return;
    const int d = (int)(idx % DINNER);
    const int t = (int)((idx / DINNER) % LSEQ);
    const int s = (int)(idx / ((size_t)DINNER * LSEQ));

    const float* base = xz2 + (size_t)s * LSEQ * (2 * DINNER) + d;
    float acc = conv_b[d];
    #pragma unroll
    for (int j = 0; j < 4; ++j) {
        const int tt = t - 3 + j;
        if (tt >= 0) acc = fmaf(conv_w[d * 4 + j],
                                base[(size_t)tt * (2 * DINNER)], acc);
    }
    xc[idx] = acc * sigmoidf_(acc);
}

// ---------------------------------------------------------------------------
// Selective scan, pass 1: per (seq, chunk, channel) compute sum(dt) and the
// local end-state h (with h0 = 0).
// ---------------------------------------------------------------------------
__global__ __launch_bounds__(128)
void scan_pass1(const float* __restrict__ dtlin, const float* __restrict__ b_dt,
                const float* __restrict__ dbl,   const float* __restrict__ xc,
                const float* __restrict__ A_log, float* __restrict__ cs)
{
    const int gid = blockIdx.x * blockDim.x + threadIdx.x;
    if (gid >= NSEQ * NCHUNK * DINNER) return;
    const int d = gid % DINNER;
    const int c = (gid / DINNER) % NCHUNK;
    const int s = gid / (DINNER * NCHUNK);

    float A[DSTATE];
    #pragma unroll
    for (int n = 0; n < DSTATE; ++n) A[n] = -__expf(A_log[d * DSTATE + n]);
    const float bdt = b_dt[d];

    float h[DSTATE];
    #pragma unroll
    for (int n = 0; n < DSTATE; ++n) h[n] = 0.0f;
    float S = 0.0f;

    for (int t = c * TCHUNK; t < (c + 1) * TCHUNK; ++t) {
        const size_t off = (size_t)s * LSEQ + t;
        const float dt  = softplusf_(dtlin[off * DINNER + d] + bdt);
        const float x   = xc[off * DINNER + d];
        const float dtx = dt * x;
        const float* Bv = dbl + off * (DTRANK + 2 * DSTATE) + DTRANK;
        S += dt;
        #pragma unroll
        for (int n = 0; n < DSTATE; ++n)
            h[n] = fmaf(h[n], __expf(dt * A[n]), dtx * Bv[n]);
    }
    float* outp = cs + (size_t)gid * (DSTATE + 1);
    #pragma unroll
    for (int n = 0; n < DSTATE; ++n) outp[n] = h[n];
    outp[DSTATE] = S;
}

// ---------------------------------------------------------------------------
// Selective scan, pass 2: rebuild h0 from prior chunks, emit y, add D-skip,
// gate with silu(z2). Writes into cols [0,384) of xz2 (z2 in cols [384,768)).
// ---------------------------------------------------------------------------
__global__ __launch_bounds__(128)
void scan_pass2(const float* __restrict__ dtlin, const float* __restrict__ b_dt,
                const float* __restrict__ dbl,   const float* __restrict__ xc,
                const float* __restrict__ A_log, const float* __restrict__ D_skip,
                const float* __restrict__ cs,    float* xz2 /* read z2 / write y */)
{
    const int gid = blockIdx.x * blockDim.x + threadIdx.x;
    if (gid >= NSEQ * NCHUNK * DINNER) return;
    const int d = gid % DINNER;
    const int c = (gid / DINNER) % NCHUNK;
    const int s = gid / (DINNER * NCHUNK);

    float A[DSTATE];
    #pragma unroll
    for (int n = 0; n < DSTATE; ++n) A[n] = -__expf(A_log[d * DSTATE + n]);
    const float bdt = b_dt[d];
    const float Dsk = D_skip[d];

    float h[DSTATE];
    #pragma unroll
    for (int n = 0; n < DSTATE; ++n) h[n] = 0.0f;
    for (int cp = 0; cp < c; ++cp) {
        const float* st =
            cs + (((size_t)(s * NCHUNK + cp)) * DINNER + d) * (DSTATE + 1);
        const float S = st[DSTATE];
        #pragma unroll
        for (int n = 0; n < DSTATE; ++n)
            h[n] = fmaf(h[n], __expf(A[n] * S), st[n]);
    }

    for (int t = c * TCHUNK; t < (c + 1) * TCHUNK; ++t) {
        const size_t off = (size_t)s * LSEQ + t;
        const float dt  = softplusf_(dtlin[off * DINNER + d] + bdt);
        const float x   = xc[off * DINNER + d];
        const float dtx = dt * x;
        const float* Bv = dbl + off * (DTRANK + 2 * DSTATE) + DTRANK;
        const float* Cv = Bv + DSTATE;
        float y = 0.0f;
        #pragma unroll
        for (int n = 0; n < DSTATE; ++n) {
            h[n] = fmaf(h[n], __expf(dt * A[n]), dtx * Bv[n]);
            y = fmaf(h[n], Cv[n], y);
        }
        const float z = xz2[off * (2 * DINNER) + DINNER + d];
        xz2[off * (2 * DINNER) + d] = (y + x * Dsk) * (z * sigmoidf_(z));
    }
}

// ---------------------------------------------------------------------------
// Merge the 4 directions (inverse permutations), gate with silu(z), write
// into cols [0,384) of xz (z read from cols [384,768)).
// ---------------------------------------------------------------------------
__global__ __launch_bounds__(256)
void merge_gate(const float* __restrict__ o, float* xz)
{
    const size_t total = (size_t)BATCH * LSEQ * DINNER;
    size_t idx = (size_t)blockIdx.x * blockDim.x + threadIdx.x;
    if (idx >= total) return;
    const int d = (int)(idx % DINNER);
    const int l = (int)((idx / DINNER) % LSEQ);
    const int b = (int)(idx / ((size_t)DINNER * LSEQ));
    const int l3 = (l & 63) * 64 + (l >> 6);

    const float f1 = o[(((size_t)(0 * BATCH + b)) * LSEQ + l)               * DINNER + d];
    const float f2 = o[(((size_t)(1 * BATCH + b)) * LSEQ + (LSEQ - 1 - l))  * DINNER + d];
    const float f3 = o[(((size_t)(2 * BATCH + b)) * LSEQ + l3)              * DINNER + d];
    const float f4 = o[(((size_t)(3 * BATCH + b)) * LSEQ + (LSEQ - 1 - l3)) * DINNER + d];

    const size_t row = (size_t)b * LSEQ + l;
    const float z = xz[row * (2 * DINNER) + DINNER + d];
    xz[row * (2 * DINNER) + d] = (f1 + f2 + f3 + f4) * (z * sigmoidf_(z));
}

// ---------------------------------------------------------------------------
extern "C" void kernel_launch(void* const* d_in, const int* in_sizes, int n_in,
                              void* d_out, int out_size, void* d_ws, size_t ws_size,
                              hipStream_t stream)
{
    const float* x      = (const float*)d_in[0];
    const float* W_in   = (const float*)d_in[1];
    const float* b_in   = (const float*)d_in[2];
    const float* Wm_in  = (const float*)d_in[3];
    const float* conv_w = (const float*)d_in[4];
    const float* conv_b = (const float*)d_in[5];
    const float* W_x    = (const float*)d_in[6];
    const float* W_dt   = (const float*)d_in[7];
    const float* b_dt   = (const float*)d_in[8];
    const float* A_log  = (const float*)d_in[9];
    const float* D_skip = (const float*)d_in[10];
    const float* Wm_out = (const float*)d_in[11];
    const float* W_out  = (const float*)d_in[12];
    const float* b_out  = (const float*)d_in[13];
    float* out = (float*)d_out;

    // workspace layout (f32 elements), with deliberate aliasing:
    float* ws    = (float*)d_ws;
    float* xz    = ws;                                            // 2*4096*768
    float* seqs  = xz    + (size_t)BATCH * LSEQ * 2 * DINNER;     // 8*4096*384 (-> xc)
    float* xz2   = seqs  + (size_t)NSEQ  * LSEQ * DINNER;         // 8*4096*768
    float* dbl   = xz2   + (size_t)NSEQ  * LSEQ * 2 * DINNER;     // 8*4096*56
    float* dtlin = dbl   + (size_t)NSEQ  * LSEQ * (DTRANK + 2 * DSTATE); // (-> o)
    float* cs    = dtlin + (size_t)NSEQ  * LSEQ * DINNER;         // 8*8*384*17

    const dim3 blk(256);
    auto grid2 = [](int M, int N) {
        return dim3((unsigned)((N + TILE_N - 1) / TILE_N),
                    (unsigned)((M + TILE_M - 1) / TILE_M));
    };
    const int MROWS = BATCH * LSEQ;   //  8192
    const int SROWS = NSEQ * LSEQ;    // 32768

    // 1) xz = x @ W_in + b_in                       (8192 x 768, K=192)
    gemm_wmma_bf16<<<grid2(MROWS, 2 * DINNER), blk, 0, stream>>>(
        x, W_in, xz, b_in, nullptr,
        MROWS, 2 * DINNER, DMODEL, DMODEL, 2 * DINNER, 2 * DINNER);

    // 2) directional sequences from xs
    {
        size_t n = (size_t)SROWS * DINNER;
        build_seqs<<<(unsigned)((n + 255) / 256), blk, 0, stream>>>(xz, seqs);
    }

    // 3) xz2 = seqs @ Wm_in                         (32768 x 768, K=384)
    gemm_wmma_bf16<<<grid2(SROWS, 2 * DINNER), blk, 0, stream>>>(
        seqs, Wm_in, xz2, nullptr, nullptr,
        SROWS, 2 * DINNER, DINNER, DINNER, 2 * DINNER, 2 * DINNER);

    // 4) depthwise causal conv + SiLU -> xc (seqs buffer reused)
    {
        size_t n = (size_t)SROWS * DINNER;
        conv_silu<<<(unsigned)((n + 255) / 256), blk, 0, stream>>>(
            xz2, conv_w, conv_b, seqs);
    }

    // 5) dbl = xc @ W_x                             (32768 x 56, K=384)
    gemm_wmma_bf16<<<grid2(SROWS, DTRANK + 2 * DSTATE), blk, 0, stream>>>(
        seqs, W_x, dbl, nullptr, nullptr,
        SROWS, DTRANK + 2 * DSTATE, DINNER,
        DINNER, DTRANK + 2 * DSTATE, DTRANK + 2 * DSTATE);

    // 6) dtlin = dt_r @ W_dt                        (32768 x 384, K=24, lda=56)
    gemm_wmma_bf16<<<grid2(SROWS, DINNER), blk, 0, stream>>>(
        dbl, W_dt, dtlin, nullptr, nullptr,
        SROWS, DINNER, DTRANK, DTRANK + 2 * DSTATE, DINNER, DINNER);

    // 7) two-pass chunked selective scan (gated output into xz2 cols 0..383)
    {
        const int nthr = NSEQ * NCHUNK * DINNER;    // 24576
        scan_pass1<<<(nthr + 127) / 128, dim3(128), 0, stream>>>(
            dtlin, b_dt, dbl, seqs, A_log, cs);
        scan_pass2<<<(nthr + 127) / 128, dim3(128), 0, stream>>>(
            dtlin, b_dt, dbl, seqs, A_log, D_skip, cs, xz2);
    }

    // 8) o = ygate @ Wm_out -> dtlin buffer         (32768 x 384, lda=768)
    gemm_wmma_bf16<<<grid2(SROWS, DINNER), blk, 0, stream>>>(
        xz2, Wm_out, dtlin, nullptr, nullptr,
        SROWS, DINNER, DINNER, 2 * DINNER, DINNER, DINNER);

    // 9) merge 4 directions + gate -> xz cols 0..383
    {
        size_t n = (size_t)MROWS * DINNER;
        merge_gate<<<(unsigned)((n + 255) / 256), blk, 0, stream>>>(dtlin, xz);
    }

    // 10) out = y2 @ W_out + b_out + x              (8192 x 192, lda=768)
    gemm_wmma_bf16<<<grid2(MROWS, DMODEL), blk, 0, stream>>>(
        xz, W_out, out, b_out, x,
        MROWS, DMODEL, DINNER, 2 * DINNER, DMODEL, DMODEL);
}
